// Rule_18090402251244
// MI455X (gfx1250) — matrix operand order repaired and checked by reading, not compiled, not run on hardware
//
#include <hip/hip_runtime.h>

// ---------------------------------------------------------------------------
// NCA update: depthwise 3x3 (fp32 VALU) + two 1x1 convs as f16 WMMA GEMMs.
//
// Roofline: 268 MB HBM traffic -> ~11.5 us floor @ 23.3 TB/s. 5.4 of 7.8
// GFLOP are 1x1 GEMMs -> v_wmma_f32_16x16x32_f16 (K=32 matches W1 exactly;
// W2[8,16] zero-padded into the same shape). Depthwise stencil stays fp32
// VALU out of an LDS halo tile.
//
// GEMM2 trick: instead of shuffling hid 8..15 from lane n+16 to lane n, keep
// each lane's GEMM1 accumulators in place and map them to K = half*16 + 0..7
// of the B operand (in this code's contiguous k order lanes 16-31 hold K=16..31). A2's columns are
// remapped to match (K 0..7 <- w2[:,0:8], K 16..23 <- w2[:,8:16], rest 0).
// Removes all ds_bpermute / cndmask glue between the two WMMAs.
// ---------------------------------------------------------------------------

typedef __attribute__((ext_vector_type(16))) _Float16 v16h;
typedef __attribute__((ext_vector_type(8)))  float    v8f;

#define Bn   16
#define Cn   8
#define Hn   512
#define Wn   512
#define HIDn 16

#define TILE       16
#define XS_STRIDE  20   // floats per halo row (18 used + pad)
#define YS_STRIDE  40   // halfs per pixel (32 used + pad); 80B keeps 16B align

union V16 {
    v16h v;
    unsigned int u[8];
    uint4 q[2];
};

static __device__ inline unsigned int pack2(float a, float b) {
    _Float16 ha = (_Float16)a, hb = (_Float16)b;
    unsigned short ua = __builtin_bit_cast(unsigned short, ha);
    unsigned short ub = __builtin_bit_cast(unsigned short, hb);
    return (unsigned int)ua | ((unsigned int)ub << 16);
}

__global__ __launch_bounds__(256)
void nca_wmma_kernel(const float* __restrict__ x, const float* __restrict__ f,
                     const float* __restrict__ w1, const float* __restrict__ b1,
                     const float* __restrict__ w2, float* __restrict__ out)
{
    __shared__ __align__(16) float    xs[Cn * 18 * XS_STRIDE];       // 11520 B halo
    __shared__ __align__(16) _Float16 ys[TILE * TILE * YS_STRIDE];   // 20480 B y (f16)

    const int tid     = threadIdx.x;     // 0..255
    const int lane    = tid & 31;
    const int wave    = tid >> 5;        // 0..7
    const int nIdx    = lane & 15;       // column (pixel) inside 16-wide group
    const int half_id = lane >> 4;       // 0: lanes 0-15, 1: lanes 16-31

    const int tx0 = blockIdx.x * TILE;
    const int ty0 = blockIdx.y * TILE;
    const int b   = blockIdx.z;

    // ---- learned 3x3 filter (uniform address -> scalarized loads) ----
    const float f00=f[0], f01=f[1], f02=f[2],
                f10=f[3], f11=f[4], f12=f[5],
                f20=f[6], f21=f[7], f22=f[8];

    // ---- A operands (16x32), contiguous k = 16*half + s order -------------------
    // lanes 0-15 : VGPR0-7 = K 0..15   (row M = lane)
    // lanes 16-31: VGPR0-7 = K 16..31  (row M = lane-16)
    V16 a1, a2;
    {
        const int m = nIdx;
        #pragma unroll
        for (int v = 0; v < 8; ++v) {
            // A1 = W1[16,32] (exact fit)
            const int kb = half_id * 16 + 2 * v;
            a1.u[v] = pack2(w1[m * 32 + kb], w1[m * 32 + kb + 1]);

            // A2: K 0..7 <- w2 cols 0..7, K 16..23 <- w2 cols 8..15,
            //     K 8..15 / 24..31 and rows >= 8 are zero.
            //     Both lane halves carry nonzero K ranges (VGPR0-3).
            float lo = 0.f, hi = 0.f;
            if (v < 4 && m < Cn) {
                const int col = half_id * 8 + 2 * v;
                lo = w2[m * 16 + col];
                hi = w2[m * 16 + col + 1];
            }
            a2.u[v] = pack2(lo, hi);
        }
    }
    // bias for the D1 rows this lane owns: hid = half_id*8 + v
    float bia[8];
    #pragma unroll
    for (int v = 0; v < 8; ++v) bia[v] = b1[half_id * 8 + v];

    // ---- stage 1: halo load (18x18 per channel, SAME zero padding) ----
    for (int i = tid; i < Cn * 18 * 18; i += 256) {
        const int c  = i / (18 * 18);
        const int r  = i % (18 * 18);
        const int hy = r / 18;
        const int hx = r % 18;
        const int gy = ty0 + hy - 1;
        const int gx = tx0 + hx - 1;
        float v = 0.f;
        if (gy >= 0 && gy < Hn && gx >= 0 && gx < Wn)
            v = x[((b * Cn + c) * Hn + gy) * Wn + gx];
        xs[(c * 18 + hy) * XS_STRIDE + hx] = v;
    }
    __syncthreads();

    // ---- stage 2: depthwise 3x3 -> ys (f16, B-matrix friendly order) ----
    {
        const int ly = tid >> 4;
        const int lx = tid & 15;
        unsigned int yu[16];               // 32 halfs, feat = 4*c + j
        #pragma unroll
        for (int c = 0; c < Cn; ++c) {
            const float* xr = &xs[(c * 18 + ly) * XS_STRIDE + lx];
            const float p00 = xr[0],             p01 = xr[1],             p02 = xr[2];
            const float p10 = xr[XS_STRIDE],     p11 = xr[XS_STRIDE + 1], p12 = xr[XS_STRIDE + 2];
            const float p20 = xr[2 * XS_STRIDE], p21 = xr[2 * XS_STRIDE + 1], p22 = xr[2 * XS_STRIDE + 2];
            const float y0 = p11;                                                     // IDENT
            const float y1 = (p02 - p00 + 2.f * (p12 - p10) + p22 - p20) * 0.125f;    // SOBEL_X/8
            const float y2 = (p00 + p02 + p20 + p22
                              + 2.f * (p01 + p10 + p12 + p21) - 12.f * p11) * 0.0625f; // LAP/16
            const float y3 = f00*p00 + f01*p01 + f02*p02
                           + f10*p10 + f11*p11 + f12*p12
                           + f20*p20 + f21*p21 + f22*p22;                             // learned
            yu[2 * c + 0] = pack2(y0, y1);
            yu[2 * c + 1] = pack2(y2, y3);
        }
        uint4* dst = (uint4*)&ys[tid * YS_STRIDE];   // 80B stride, 16B aligned
        const uint4* src = (const uint4*)yu;
        dst[0] = src[0]; dst[1] = src[1]; dst[2] = src[2]; dst[3] = src[3];
    }
    __syncthreads();

    // ---- stage 3: per-wave WMMA GEMMs over two 16-pixel rows ----------
    // B operand as loaded here (16-bit, 32x16, contiguous k): lane = column; lanes 0-15 hold K 0..15
    // contiguously (2 per VGPR), lanes 16-31 hold K 16..31.
    #pragma unroll
    for (int g = 0; g < 2; ++g) {
        const int row = wave * 2 + g;         // tile row 0..15
        const int pix = row * 16 + nIdx;      // local pixel id

        V16 bm;
        const uint4* src = (const uint4*)&ys[pix * YS_STRIDE + half_id * 16];
        bm.q[0] = src[0];
        bm.q[1] = src[1];

        v8f acc = {};
        acc = __builtin_amdgcn_wmma_f32_16x16x32_f16(
                  false, a1.v, false, bm.v, (short)0, acc, false, false);

        // bias + relu in f32; D1: lane owns hid = half_id*8 + v for pixel nIdx
        float hv[8];
        #pragma unroll
        for (int v = 0; v < 8; ++v) {
            const float t = acc[v] + bia[v];
            hv[v] = t > 0.f ? t : 0.f;
        }

        // B2: each lane's own h values go to K = half_id*16 + 0..7;
        // K = half_id*16 + 8..15 are zero. No cross-lane movement needed.
        V16 b2;
        #pragma unroll
        for (int j = 0; j < 4; ++j)
            b2.u[j] = pack2(hv[2 * j], hv[2 * j + 1]);
        #pragma unroll
        for (int j = 4; j < 8; ++j)
            b2.u[j] = 0u;

        v8f acc2 = {};
        acc2 = __builtin_amdgcn_wmma_f32_16x16x32_f16(
                   false, a2.v, false, b2.v, (short)0, acc2, false, false);

        // D2: lane l, VGPR v -> ch = half_id*8 + v, pixel nIdx.
        // Only ch < 8 is real (A2 rows 8-15 are zero) -> lanes 0-15 store.
        if (half_id == 0) {
            const int gy = ty0 + row;
            const int gx = tx0 + nIdx;
            #pragma unroll
            for (int v = 0; v < 8; ++v) {
                const float xv = xs[(v * 18 + (row + 1)) * XS_STRIDE + (nIdx + 1)];
                out[((b * Cn + v) * Hn + gy) * Wn + gx] = xv + acc2[v];
            }
        }
    }
}

extern "C" void kernel_launch(void* const* d_in, const int* in_sizes, int n_in,
                              void* d_out, int out_size, void* d_ws, size_t ws_size,
                              hipStream_t stream) {
    const float* x  = (const float*)d_in[0];   // [16,8,512,512]
    const float* f  = (const float*)d_in[1];   // [3,3]
    const float* w1 = (const float*)d_in[2];   // [16,32]
    const float* b1 = (const float*)d_in[3];   // [16]
    const float* w2 = (const float*)d_in[4];   // [8,16]
    float* out = (float*)d_out;                // [16,8,512,512]

    dim3 grid(Wn / TILE, Hn / TILE, Bn);       // (32, 32, 16) = 16384 WGs
    nca_wmma_kernel<<<grid, 256, 0, stream>>>(x, f, w1, b1, w2, out);
}
